// SelfAttention1D_52733608460595
// MI455X (gfx1250) — matrix operand order repaired and checked by reading, not compiled, not run on hardware
//
#include <hip/hip_runtime.h>
#include <hip/hip_bf16.h>
#include <math.h>

#define DEVFN __device__ __forceinline__

typedef __attribute__((ext_vector_type(16))) __bf16 v16bf;
typedef __attribute__((ext_vector_type(8)))  __bf16 v8bf;
typedef __attribute__((ext_vector_type(8)))  float  v8f;
typedef __attribute__((ext_vector_type(4)))  float  v4f;

constexpr int Bb = 4;     // batch
constexpr int Cc = 128;   // channels
constexpr int Aa = 64;    // attention dim
constexpr int Nn = 4096;  // sequence length

// ---- fragment helpers -------------------------------------------------------
// 16 contiguous bf16 (32B) as two 16B loads (16B alignment guaranteed by layout)
DEVFN v16bf ld16c(const __bf16* p) {
  v8bf lo = *(const v8bf*)p;
  v8bf hi = *(const v8bf*)(p + 8);
  v16bf r;
#pragma unroll
  for (int i = 0; i < 8; ++i) { r[i] = lo[i]; r[i + 8] = hi[i]; }
  return r;
}
// two disjoint 8-element (16B) chunks -> A-matrix fragment halves
DEVFN v16bf ld8x2(const __bf16* p0, const __bf16* p1) {
  v8bf lo = *(const v8bf*)p0;
  v8bf hi = *(const v8bf*)p1;
  v16bf r;
#pragma unroll
  for (int i = 0; i < 8; ++i) { r[i] = lo[i]; r[i + 8] = hi[i]; }
  return r;
}

DEVFN v8f wmma_bf16(v16bf a, v16bf b, v8f c) {
  // D = A(16x32) * B(32x16) + C, f32 accumulate
  return __builtin_amdgcn_wmma_f32_16x16x32_bf16(
      /*neg_a=*/false, a, /*neg_b=*/false, b,
      /*c_mod=*/(short)0, c, /*reuse_a=*/false, /*reuse_b=*/false);
}

// reductions across the 16-lane half-wave groups (wave32)
DEVFN float rmax16(float v) {
  v = fmaxf(v, __shfl_xor(v, 1, 32));
  v = fmaxf(v, __shfl_xor(v, 2, 32));
  v = fmaxf(v, __shfl_xor(v, 4, 32));
  v = fmaxf(v, __shfl_xor(v, 8, 32));
  return v;
}
DEVFN float rsum16(float v) {
  v += __shfl_xor(v, 1, 32);
  v += __shfl_xor(v, 2, 32);
  v += __shfl_xor(v, 4, 32);
  v += __shfl_xor(v, 8, 32);
  return v;
}

// ---- kernel 1: q/k/v projections via WMMA ----------------------------------
// Writes: qT,kT as [B][N][A] bf16 (contiguous along A), v as [B][C][N] bf16.
__global__ __launch_bounds__(256) void proj_kernel(
    const float* __restrict__ x,
    const float* __restrict__ Wq, const float* __restrict__ bq,
    const float* __restrict__ Wk, const float* __restrict__ bk,
    const float* __restrict__ Wv, const float* __restrict__ bv,
    __bf16* __restrict__ qT, __bf16* __restrict__ kT, __bf16* __restrict__ vW) {
  constexpr int LDP = 136;  // 128 + 8 pad (row stride 272B: bank-conflict break)
  __shared__ __align__(16) __bf16 xs[64 * LDP];  // x^T tile: [n][c] bf16

  const int b  = blockIdx.x / (Nn / 64);
  const int n0 = (blockIdx.x % (Nn / 64)) * 64;
  const int tid = threadIdx.x;

  // stage x^T tile: 64 n x 128 c, coalesced along n
  for (int i = tid; i < 64 * 128; i += 256) {
    const int n = i & 63, c = i >> 6;
    xs[n * LDP + c] = (__bf16)x[(size_t)b * Cc * Nn + (size_t)c * Nn + n0 + n];
  }
  __syncthreads();

  const int wave = tid >> 5, lane = tid & 31;
  const int hl = lane >> 4;       // half-wave select (0/1)
  const int l15 = lane & 15;

  // 64 output tiles: 0..15 q, 16..31 k, 32..63 v; 8 waves round-robin
  for (int t = wave; t < 64; t += 8) {
    v8f accv = {};
    if (t < 32) {
      // ---- q or k: D[n][a] tile, K = c (128) ----
      const float* W    = (t < 16) ? Wq : Wk;
      const float* bias = (t < 16) ? bq : bk;
      __bf16* dst       = (t < 16) ? qT : kT;
      const int tt = t & 15, tn = tt >> 2, ta = tt & 3;
#pragma unroll
      for (int kk = 0; kk < 4; ++kk) {
        // A = x^T rows (M=n, K=c) from LDS
        const __bf16* ap = &xs[(tn * 16 + l15) * LDP + kk * 32 + hl * 16];
        v16bf af = ld8x2(ap, ap + 8);
        // B = W^T (K=c, N=a): W[a][c] contiguous along c; f32 -> bf16 on the fly
        const v4f* wp = (const v4f*)(W + (size_t)(ta * 16 + l15) * Cc + kk * 32 + hl * 16);
        v4f w0 = wp[0], w1 = wp[1], w2 = wp[2], w3 = wp[3];
        v16bf bfv;
#pragma unroll
        for (int e = 0; e < 4; ++e) {
          bfv[e]      = (__bf16)w0[e];
          bfv[e + 4]  = (__bf16)w1[e];
          bfv[e + 8]  = (__bf16)w2[e];
          bfv[e + 12] = (__bf16)w3[e];
        }
        accv = wmma_bf16(af, bfv, accv);
      }
      const float bias_l = bias[ta * 16 + l15];  // N dim = a across lanes
#pragma unroll
      for (int r = 0; r < 8; ++r) {
        const int n = tn * 16 + r + hl * 8;
        dst[((size_t)b * Nn + n0 + n) * Aa + ta * 16 + l15] = (__bf16)(accv[r] + bias_l);
      }
    } else {
      // ---- v: D[c'][n] tile, K = c (128) ----
      const int tt = t - 32, tc = tt >> 2, tn = tt & 3;
#pragma unroll
      for (int kk = 0; kk < 4; ++kk) {
        // A = Wv rows (M=c', K=c): f32 -> bf16
        const float* ap0 = Wv + (size_t)(tc * 16 + l15) * Cc + kk * 32 + hl * 16;
        const v4f* wp0 = (const v4f*)ap0;
        const v4f* wp1 = (const v4f*)(ap0 + 8);
        v4f a0 = wp0[0], a1 = wp0[1], a2 = wp1[0], a3 = wp1[1];
        v16bf af;
#pragma unroll
        for (int e = 0; e < 4; ++e) {
          af[e]      = (__bf16)a0[e];
          af[e + 4]  = (__bf16)a1[e];
          af[e + 8]  = (__bf16)a2[e];
          af[e + 12] = (__bf16)a3[e];
        }
        // B = x (K=c, N=n): x^T[n][c] contiguous along c in LDS
        const __bf16* bp = &xs[(tn * 16 + l15) * LDP + kk * 32 + hl * 16];
        v16bf bfv = ld16c(bp);
        accv = wmma_bf16(af, bfv, accv);
      }
#pragma unroll
      for (int r = 0; r < 8; ++r) {
        const int cp = tc * 16 + r + hl * 8;  // M dim = c' across VGPRs/halves
        vW[((size_t)b * Cc + cp) * Nn + n0 + tn * 16 + l15] =
            (__bf16)(accv[r] + bv[cp]);
      }
    }
  }
}

// ---- kernel 2: flash attention + residual ----------------------------------
// 2 waves/WG (512 WGs for spread); wave owns 16 query rows; 64-key blocks.
__global__ __launch_bounds__(64) void attn_kernel(
    const float* __restrict__ x,
    const __bf16* __restrict__ qT, const __bf16* __restrict__ kT,
    const __bf16* __restrict__ vW, float* __restrict__ out) {
  constexpr int LDPP = 72;  // 64 + 8 pad (row stride 144B)
  constexpr int LDO  = 17;  // 16 + 1 pad f32 for epilogue transpose
  __shared__ __align__(16) __bf16 pl[2 * 16 * LDPP];    // per-wave P tiles
  __shared__ __align__(16) float  ctxs[2 * Cc * LDO];   // per-wave ctx transpose

  const int b  = blockIdx.x / (Nn / 32);
  const int n0 = (blockIdx.x % (Nn / 32)) * 32;
  const int wave = threadIdx.x >> 5, lane = threadIdx.x & 31;
  const int hl = lane >> 4, l15 = lane & 15;
  const int nrow = n0 + wave * 16;
  __bf16* pw = &pl[wave * 16 * LDPP];

  // Q A-fragments (M=n, K=a): loaded once, reused for all key blocks
  const __bf16* qp = qT + ((size_t)b * Nn + nrow + l15) * Aa;
  const v16bf q0 = ld8x2(qp + hl * 16,      qp + 8 + hl * 16);  // K = a 0..31
  const v16bf q1 = ld8x2(qp + 32 + hl * 16, qp + 40 + hl * 16);  // K = a 32..63

  v8f acc[8];
  const v8f vzero = {};
#pragma unroll
  for (int i = 0; i < 8; ++i) acc[i] = vzero;
  float m8[8], l8[8];
#pragma unroll
  for (int r = 0; r < 8; ++r) { m8[r] = -__builtin_inff(); l8[r] = 0.0f; }

  const __bf16* kbase = kT + (size_t)b * Nn * Aa;
  const __bf16* vbase = vW + (size_t)b * Cc * Nn;

  for (int m0 = 0; m0 < Nn; m0 += 64) {
    if (m0 + 64 < Nn) {  // global_prefetch_b8 of next key/value blocks
      __builtin_prefetch((const void*)(kbase + (size_t)(m0 + 64 + l15) * Aa), 0, 1);
      __builtin_prefetch((const void*)(vbase + (size_t)l15 * Nn + m0 + 64), 0, 1);
    }

    // ---- scores: 4 subtiles of 16 keys, K = a (64) ----
    v8f S[4];
#pragma unroll
    for (int s = 0; s < 4; ++s) {
      const __bf16* kp = kbase + (size_t)(m0 + s * 16 + l15) * Aa + hl * 16;
      v16bf klo = ld16c(kp);        // K = a 0..31 chunk (B: K contiguous)
      v16bf khi = ld16c(kp + 32);   // K = a 32..63 chunk
      S[s] = wmma_bf16(q0, klo, vzero);
      S[s] = wmma_bf16(q1, khi, S[s]);
    }

    // ---- online softmax per row, amortized over 64 keys ----
#pragma unroll
    for (int r = 0; r < 8; ++r) {
      float cm = fmaxf(fmaxf(S[0][r], S[1][r]), fmaxf(S[2][r], S[3][r]));
      cm = rmax16(cm);
      const float mn = fmaxf(m8[r], cm);
      const float sc = __expf(m8[r] - mn);
      m8[r] = mn;
      float rs = 0.0f;
#pragma unroll
      for (int s = 0; s < 4; ++s) {
        const float p = __expf(S[s][r] - mn);
        S[s][r] = p;
        rs += p;
      }
      l8[r] = l8[r] * sc + rsum16(rs);
#pragma unroll
      for (int ct = 0; ct < 8; ++ct) acc[ct][r] *= sc;
    }

    // ---- C/D layout -> A-matrix layout for P via per-wave LDS ----
#pragma unroll
    for (int r = 0; r < 8; ++r) {
      const int row = r + hl * 8;
#pragma unroll
      for (int s = 0; s < 4; ++s)
        pw[row * LDPP + s * 16 + l15] = (__bf16)S[s][r];
    }
    const __bf16* pp = &pw[l15 * LDPP];
    v16bf pf0 = ld8x2(pp + hl * 16,      pp + 8 + hl * 16);  // K = key 0..31
    v16bf pf1 = ld8x2(pp + 32 + hl * 16, pp + 40 + hl * 16);  // K = key 32..63

    // ---- ctx += P(16x64) * v^T(64x16) for 8 channel tiles ----
#pragma unroll
    for (int ct = 0; ct < 8; ++ct) {
      const __bf16* vp = vbase + (size_t)(ct * 16 + l15) * Nn + m0 + hl * 16;
      v16bf v0 = ld16c(vp);         // B: K = key contiguous, N = channel lanes
      v16bf v1 = ld16c(vp + 32);
      acc[ct] = wmma_bf16(pf0, v0, acc[ct]);
      acc[ct] = wmma_bf16(pf1, v1, acc[ct]);
    }
  }

  // ---- epilogue: normalize, transpose via LDS, coalesced residual+store ----
  float rinv[8];
#pragma unroll
  for (int r = 0; r < 8; ++r) rinv[r] = 1.0f / l8[r];

  float* cw = &ctxs[wave * Cc * LDO];  // [c][n] f32, padded rows
#pragma unroll
  for (int ct = 0; ct < 8; ++ct) {
#pragma unroll
    for (int r = 0; r < 8; ++r) {
      const int c = ct * 16 + l15;       // N dim = channel across lanes
      const int n = r + hl * 8;          // M dim = query row
      cw[c * LDO + n] = acc[ct][r] * rinv[r];
    }
  }
  // per-wave data only: in-order DS ops within the wave, no barrier needed
  for (int i = lane; i < Cc * 16; i += 32) {
    const int c = i >> 4, n = i & 15;
    const size_t idx = ((size_t)b * Cc + c) * Nn + nrow + n;
    out[idx] = x[idx] + cw[c * LDO + n];  // lanes run along n: coalesced
  }
}

// ---- launcher ---------------------------------------------------------------
extern "C" void kernel_launch(void* const* d_in, const int* in_sizes, int n_in,
                              void* d_out, int out_size, void* d_ws, size_t ws_size,
                              hipStream_t stream) {
  (void)in_sizes; (void)n_in; (void)out_size; (void)ws_size;
  const float* x  = (const float*)d_in[0];
  const float* Wq = (const float*)d_in[1];
  const float* bq = (const float*)d_in[2];
  const float* Wk = (const float*)d_in[3];
  const float* bk = (const float*)d_in[4];
  const float* Wv = (const float*)d_in[5];
  const float* bv = (const float*)d_in[6];
  float* out = (float*)d_out;

  // workspace: qT [B][N][A] bf16 | kT [B][N][A] bf16 | v [B][C][N] bf16 = 8 MB
  char* ws = (char*)d_ws;
  __bf16* qT = (__bf16*)ws;
  __bf16* kT = (__bf16*)(ws + (size_t)Bb * Nn * Aa * 2);
  __bf16* vW = (__bf16*)(ws + (size_t)2 * Bb * Nn * Aa * 2);

  proj_kernel<<<dim3(Bb * (Nn / 64)), 256, 0, stream>>>(x, Wq, bq, Wk, bk, Wv, bv,
                                                        qT, kT, vW);
  attn_kernel<<<dim3(Bb * (Nn / 32)), 64, 0, stream>>>(x, qT, kT, vW, out);
}
